// NetworksPlusCircuit_22033182228737
// MI455X (gfx1250) — matrix operand order repaired and checked by reading, not compiled, not run on hardware
//
#include <hip/hip_runtime.h>

// ---------------- problem constants ----------------
#define BATCH 65536
#define KDIM  784      // 28*28
#define HID   512
#define NCLS  10
#define NSUM  19

// ---------------- tiling ----------------
#define MT    64       // batch rows per workgroup
#define NT    128      // hidden cols per N-chunk (4 chunks cover 512)
#define KC    64       // K chunk (2 WMMA k-steps of 32)
#define NKC   13       // ceil(784/64) -> K padded to 832
#define KPAD  (NKC*KC) // 832
#define XSTR  840      // sX row stride (1680B, 16B aligned, bank-spread)
#define BSTR  72       // sB row stride (144B, 16B aligned, bank-spread)
#define HSTR  520      // sH row stride

typedef __attribute__((ext_vector_type(16))) __bf16 v16bf;
typedef __attribute__((ext_vector_type(8)))  __bf16 v8bf;
typedef __attribute__((ext_vector_type(4)))  __bf16 v4bf;
typedef __attribute__((ext_vector_type(8)))  float  v8f;
typedef __attribute__((ext_vector_type(4)))  float  v4f;

__device__ __forceinline__ v16bf cat8(v8bf lo, v8bf hi) {
  return __builtin_shufflevector(lo, hi, 0,1,2,3,4,5,6,7,8,9,10,11,12,13,14,15);
}

// In-flight registers for one thread's 4(K)x8(N) piece of a W1 chunk.
struct WRegs { float v[4][8]; };

// Load half: issue global reads for one 64(K)x128(N) W1 chunk (no LDS traffic).
__device__ __forceinline__ void load_w_chunk(WRegs& w, const float* __restrict__ W1,
                                             int kbase, int nbase, int tid) {
  const int kk = (tid >> 4) * 4;   // 0..60
  const int n0 = (tid & 15) * 8;   // 0..120
  #pragma unroll
  for (int r = 0; r < 4; ++r) {
    const int gk = kbase + kk + r;
    if (gk < KDIM) {
      const float* src = W1 + (size_t)gk * HID + nbase + n0;
      v4f a = *(const v4f*)(src);
      v4f b = *(const v4f*)(src + 4);
      #pragma unroll
      for (int i = 0; i < 4; ++i) { w.v[r][i] = a[i]; w.v[r][i + 4] = b[i]; }
    } else {
      #pragma unroll
      for (int i = 0; i < 8; ++i) w.v[r][i] = 0.0f;
    }
  }
}

// Store half: f32->bf16 convert and write transposed (n-major), b64-packed.
__device__ __forceinline__ void store_w_chunk(__bf16 (*sBbuf)[BSTR], const WRegs& w, int tid) {
  const int kk = (tid >> 4) * 4;
  const int n0 = (tid & 15) * 8;
  #pragma unroll
  for (int n = 0; n < 8; ++n) {
    v4bf p;
    #pragma unroll
    for (int r = 0; r < 4; ++r) p[r] = (__bf16)w.v[r][n];
    *(v4bf*)&sBbuf[n0 + n][kk] = p;   // transposed, 8B packed store
  }
}

__global__ __launch_bounds__(256)
void fused_mlp_sdd_kernel(const float* __restrict__ x,
                          const float* __restrict__ W1_0, const float* __restrict__ b1_0,
                          const float* __restrict__ W2_0, const float* __restrict__ b2_0,
                          const float* __restrict__ W1_1, const float* __restrict__ b1_1,
                          const float* __restrict__ W2_1, const float* __restrict__ b2_1,
                          float* __restrict__ out)
{
  __shared__ __align__(16) __bf16 sX[MT][XSTR];      // resident x tile (bf16, K-padded)
  __shared__ __align__(16) __bf16 sB[2][NT][BSTR];   // W1 chunk, transposed, double-buffered
  __shared__ __align__(16) __bf16 sH[MT][HSTR];      // relu(x@W1+b1)
  __shared__ __align__(16) float  sW2[HID][NCLS];    // W2 cached
  __shared__ __align__(16) float  sLog[MT][NCLS];    // logits
  __shared__ __align__(16) float  sP[2][MT][NCLS];   // softmax probs, both nets

  const int tid  = threadIdx.x;
  const int lane = tid & 31;
  const int wave = tid >> 5;
  const int row0 = blockIdx.x * MT;

  // wave tile inside the 64x128 C tile: 4 (M) x 2 (N) wave grid
  const int wm = (wave & 3) * 16;   // wave row base
  const int wn = (wave >> 2) * 64;  // wave col base within NT
  const int lm = lane & 15;
  const int lh = lane >> 4;

  for (int net = 0; net < 2; ++net) {
    const float* __restrict__ W1 = net ? W1_1 : W1_0;
    const float* __restrict__ b1 = net ? b1_1 : b1_0;
    const float* __restrict__ W2 = net ? W2_1 : W2_0;
    const float* __restrict__ b2 = net ? b2_1 : b2_0;
    const float* __restrict__ xn = x + (size_t)net * BATCH * KDIM;

    // ---- Phase 1a: x rows -> sX (f32 -> bf16, zero-pad K 784..831) ----
    // 64 rows * 52 groups of 16 = 3328 tasks = 13 per thread, coalesced 64B/thread
    #pragma unroll 1
    for (int j = 0; j < 13; ++j) {
      const int task = j * 256 + tid;
      const int m  = task / 52;
      const int cg = task - m * 52;
      const int k  = cg * 16;
      const float* src = xn + (size_t)(row0 + m) * KDIM + k;
      v8bf lo, hi;
      if (k + 16 <= KDIM) {
        v4f a0 = *(const v4f*)(src);
        v4f a1 = *(const v4f*)(src + 4);
        v4f a2 = *(const v4f*)(src + 8);
        v4f a3 = *(const v4f*)(src + 12);
        #pragma unroll
        for (int i = 0; i < 4; ++i) {
          lo[i]   = (__bf16)a0[i];  lo[i+4] = (__bf16)a1[i];
          hi[i]   = (__bf16)a2[i];  hi[i+4] = (__bf16)a3[i];
        }
      } else {
        #pragma unroll
        for (int i = 0; i < 8; ++i) {
          float f0 = (k + i     < KDIM) ? src[i]     : 0.0f;
          float f1 = (k + 8 + i < KDIM) ? src[8 + i] : 0.0f;
          lo[i] = (__bf16)f0;  hi[i] = (__bf16)f1;
        }
      }
      *(v8bf*)&sX[m][k]     = lo;
      *(v8bf*)&sX[m][k + 8] = hi;
    }

    // ---- Phase 1a': W2 -> LDS (5120 floats / 256 = 20 each, coalesced) ----
    #pragma unroll 1
    for (int j = 0; j < 20; ++j) {
      const int idx = j * 256 + tid;
      (&sW2[0][0])[idx] = W2[idx];
    }
    __syncthreads();

    // ---- Phase 1b: H = relu(x @ W1 + b1), N swept in 4 chunks of 128 ----
    #pragma unroll 1
    for (int nc = 0; nc < 4; ++nc) {
      v8f acc[4];
      #pragma unroll
      for (int cb = 0; cb < 4; ++cb)
        acc[cb] = (v8f){0.f,0.f,0.f,0.f,0.f,0.f,0.f,0.f};

      // prologue: stage chunk 0 into buffer 0
      {
        WRegs w0;
        load_w_chunk(w0, W1, 0, nc * NT, tid);
        store_w_chunk(sB[0], w0, tid);
      }
      __syncthreads();

      #pragma unroll 1
      for (int kc = 0; kc < NKC; ++kc) {
        const int buf   = kc & 1;
        const int kbase = kc * KC;

        // issue global loads for the next chunk NOW; no wait until after the WMMAs
        WRegs wnext;
        if (kc + 1 < NKC)
          load_w_chunk(wnext, W1, kbase + KC, nc * NT, tid);

        // hint chunk kc+2 into cache (global_prefetch_b8)
        if (kc + 2 < NKC) {
          const int pk = kbase + 2 * KC + (tid >> 3);
          if (pk < KDIM)
            __builtin_prefetch(W1 + (size_t)pk * HID + nc * NT + (tid & 7) * 16, 0, 0);
        }

        // 2 WMMA k-steps of 32 over this 64-wide K chunk
        #pragma unroll
        for (int ks = 0; ks < 2; ++ks) {
          // A frag: 16-bit A layout — lh=0: K 0..7 & 16..23; lh=1: 8..15 & 24..31
          const int ka = kbase + ks * 32 + lh * 8;
          v16bf afrag = cat8(*(const v8bf*)&sX[wm + lm][ka],
                             *(const v8bf*)&sX[wm + lm][ka + 16]);
          #pragma unroll
          for (int cb = 0; cb < 4; ++cb) {
            // B frag: 16-bit B layout — lane n, lh=0: K 0..7 & 16..23; lh=1: 8..15 & 24..31
            const int bn = wn + cb * 16 + lm;
            const int kb = ks * 32 + lh * 8;
            v16bf bfrag = cat8(*(const v8bf*)&sB[buf][bn][kb],
                               *(const v8bf*)&sB[buf][bn][kb + 16]);
            acc[cb] = __builtin_amdgcn_wmma_f32_16x16x32_bf16(
                false, afrag, false, bfrag, (short)0, acc[cb], false, false);
          }
        }

        // store half: convert + LDS-write next chunk into the other buffer.
        // Safe: all reads of sB[buf^1] finished at the previous barrier.
        if (kc + 1 < NKC)
          store_w_chunk(sB[buf ^ 1], wnext, tid);

        __syncthreads();   // next-chunk stores visible; current buffer free for kc+2
      }

      // epilogue: +b1, ReLU, -> sH (bf16). C layout: reg i -> row i+8*lh, col lm
      #pragma unroll
      for (int cb = 0; cb < 4; ++cb) {
        const int gcol = nc * NT + wn + cb * 16 + lm;
        const float bias = b1[gcol];
        #pragma unroll
        for (int i = 0; i < 8; ++i) {
          const int r = wm + lh * 8 + i;
          float h = acc[cb][i] + bias;
          sH[r][gcol] = (__bf16)(h > 0.0f ? h : 0.0f);
        }
      }
      __syncthreads();
    }

    // ---- Phase 2: logits = H @ W2 + b2 (640 row,class dot products) ----
    #pragma unroll 1
    for (int task = tid; task < MT * NCLS; task += 256) {
      const int r = task / NCLS;
      const int c = task - r * NCLS;
      float s = 0.0f;
      #pragma unroll 8
      for (int k = 0; k < HID; ++k)
        s += (float)sH[r][k] * sW2[k][c];
      sLog[r][c] = s + b2[c];
    }
    __syncthreads();

    // ---- softmax per row ----
    if (tid < MT) {
      const int r = tid;
      float mx = sLog[r][0];
      #pragma unroll
      for (int c = 1; c < NCLS; ++c) mx = fmaxf(mx, sLog[r][c]);
      float e[NCLS];
      float s = 0.0f;
      #pragma unroll
      for (int c = 0; c < NCLS; ++c) { e[c] = __expf(sLog[r][c] - mx); s += e[c]; }
      const float inv = 1.0f / s;
      #pragma unroll
      for (int c = 0; c < NCLS; ++c) sP[net][r][c] = e[c] * inv;
    }
    __syncthreads();
  }

  // ---- Phase 3: out[n][s] = sum_{a+c=s} p0[n][a] * p1[n][c] ----
  if (tid < MT) {
    const int r = tid;
    float p0[NCLS], p1[NCLS];
    #pragma unroll
    for (int c = 0; c < NCLS; ++c) { p0[c] = sP[0][r][c]; p1[c] = sP[1][r][c]; }
    float accv[NSUM];
    #pragma unroll
    for (int s = 0; s < NSUM; ++s) accv[s] = 0.0f;
    #pragma unroll
    for (int a = 0; a < NCLS; ++a)
      #pragma unroll
      for (int c = 0; c < NCLS; ++c)
        accv[a + c] = fmaf(p0[a], p1[c], accv[a + c]);
    float* dst = out + (size_t)(row0 + r) * NSUM;
    #pragma unroll
    for (int s = 0; s < NSUM; ++s) dst[s] = accv[s];
  }
}

extern "C" void kernel_launch(void* const* d_in, const int* in_sizes, int n_in,
                              void* d_out, int out_size, void* d_ws, size_t ws_size,
                              hipStream_t stream) {
  const float* x    = (const float*)d_in[0];
  const float* W1_0 = (const float*)d_in[1];
  const float* b1_0 = (const float*)d_in[2];
  const float* W2_0 = (const float*)d_in[3];
  const float* b2_0 = (const float*)d_in[4];
  const float* W1_1 = (const float*)d_in[5];
  const float* b1_1 = (const float*)d_in[6];
  const float* W2_1 = (const float*)d_in[7];
  const float* b2_1 = (const float*)d_in[8];
  float* out = (float*)d_out;

  dim3 grid(BATCH / MT);   // 1024 workgroups
  dim3 block(256);         // 8 wave32 waves
  hipLaunchKernelGGL(fused_mlp_sdd_kernel, grid, block, 0, stream,
                     x, W1_0, b1_0, W2_0, b2_0, W1_1, b1_1, W2_1, b2_1, out);
}
